// ContinuousConvolutionModel_76441827934480
// MI455X (gfx1250) — matrix operand order repaired and checked by reading, not compiled, not run on hardware
//
#include <hip/hip_runtime.h>

typedef __attribute__((ext_vector_type(16))) _Float16 v16h;
typedef __attribute__((ext_vector_type(8)))  float    v8f;

#define N_PTS  6000
#define K_NBR  16
#define CELLS  216
#define SLOTS  128              // K_NBR * 8 corners
#define TILE   32               // points per block (two 16-row WMMA sub-tiles)
#define NTILES ((N_PTS + TILE - 1) / TILE)   // 188 (last tile has 16 valid points)
#define CCH    4                // kernel cells per LDS chunk (216 = 54 * 4)

__device__ __forceinline__ int iclamp(int v, int lo, int hi) {
    return v < lo ? lo : (v > hi ? hi : v);
}

// ---------------------------------------------------------------------------
// Geometry precompute: for every (point, neighbor) compute the 8 trilinear
// (cell index, weight) slots. Depends only on pos/nbr/mask -> run once,
// reused by all 5 layers.
// ---------------------------------------------------------------------------
__global__ void k_slots(const float* __restrict__ pos,
                        const int* __restrict__ nbr,
                        const float* __restrict__ mask,
                        unsigned short* __restrict__ sIdx,
                        float* __restrict__ sW) {
    int t = blockIdx.x * blockDim.x + threadIdx.x;
    if (t >= N_PTS * K_NBR) return;
    int n  = t / K_NBR;
    int nb = nbr[t];
    float mk = mask[t];

    float x = (pos[nb*3+0] - pos[n*3+0]) * (2.0f/3.0f);   // 2/EXTENT
    float y = (pos[nb*3+1] - pos[n*3+1]) * (2.0f/3.0f);
    float z = (pos[nb*3+2] - pos[n*3+2]) * (2.0f/3.0f);

    // ---- map ball to cube (matches reference _map_ball_to_cube) ----
    float sq     = x*x + y*y + z*z;
    bool  inside = sq > 1e-10f;
    float norm   = sqrtf(inside ? sq : 1.0f);
    bool  pole   = 1.25f*z*z > x*x + y*y;
    float s_p    = sqrtf(3.0f*norm / (norm + fabsf(z)));
    float rxy_sq = x*x + y*y;
    float rxy    = sqrtf((pole || !inside) ? 1.0f : rxy_sq);
    float s_s    = norm / rxy;
    float sgnz   = (z > 0.f) ? 1.f : ((z < 0.f) ? -1.f : 0.f);
    float cx     = inside ? (pole ? x*s_p : x*s_s) : 0.f;
    float cy     = inside ? (pole ? y*s_p : y*s_s) : 0.f;
    float cz     = inside ? (pole ? sgnz*norm : 1.5f*z) : 0.f;
    float rc_sq  = cx*cx + cy*cy;
    bool  rin    = rc_sq > 1e-10f;
    float rn     = sqrtf(rin ? rc_sq : 1.0f);
    bool  condx  = fabsf(cx) >= fabsf(cy);
    float den0   = condx ? cx : cy;
    float den    = (fabsf(den0) > 1e-10f) ? den0 : 1.0f;
    float tt     = (4.0f/3.14159265358979323846f) * atanf((condx ? cy : cx) / den);
    float major  = (den > 0.f) ? rn : -rn;
    float u = rin ? (condx ? major      : major*tt) : 0.f;
    float v = rin ? (condx ? major*tt   : major   ) : 0.f;

    // ---- trilinear corner slots ----
    float m3[3] = {u, v, cz};
    int   i0[3]; float fr[3];
    #pragma unroll
    for (int d = 0; d < 3; ++d) {
        float tq = (m3[d]*0.5f + 0.5f)*6.0f - 0.5f;
        float fl = floorf(tq);
        i0[d] = (int)fl;
        fr[d] = tq - fl;
    }
    int   ix[2] = { iclamp(i0[0],0,5), iclamp(i0[0]+1,0,5) };
    int   iy[2] = { iclamp(i0[1],0,5), iclamp(i0[1]+1,0,5) };
    int   iz[2] = { iclamp(i0[2],0,5), iclamp(i0[2]+1,0,5) };
    float wx[2] = { 1.f-fr[0], fr[0] };
    float wy[2] = { 1.f-fr[1], fr[1] };
    float wz[2] = { 1.f-fr[2], fr[2] };

    int base = t * 8;
    #pragma unroll
    for (int a = 0; a < 2; ++a)
        #pragma unroll
        for (int b = 0; b < 2; ++b)
            #pragma unroll
            for (int c = 0; c < 2; ++c) {
                int s = a*4 + b*2 + c;
                sIdx[base + s] = (unsigned short)(ix[a]*36 + iy[b]*6 + iz[c]);
                sW[base + s]   = wx[a]*wy[b]*wz[c]*mk;
            }
}

// ---------------------------------------------------------------------------
// Weight prep: W (216, cin, cout) f32 -> Wt (216, coutP, cinP) f16, zero pad.
// Layout puts c fastest so each WMMA B-fragment lane reads 16 contiguous halves.
// ---------------------------------------------------------------------------
__global__ void k_prepw(const float* __restrict__ W, _Float16* __restrict__ Wt,
                        int cin, int cout, int cinP, int coutP) {
    int i = blockIdx.x * blockDim.x + threadIdx.x;
    int total = CELLS * coutP * cinP;
    if (i >= total) return;
    int c    = i % cinP;
    int o    = (i / cinP) % coutP;
    int cell = i / (cinP * coutP);
    float v = (c < cin && o < cout) ? W[((size_t)cell*cin + c)*cout + o] : 0.f;
    Wt[i] = (_Float16)v;
}

__global__ void k_prepb(const float* __restrict__ b, float* __restrict__ bp,
                        int cout, int coutP) {
    int i = threadIdx.x;
    if (i < coutP) bp[i] = (i < cout) ? b[i] : 0.f;
}

__global__ void k_castfeats(const float* __restrict__ f, _Float16* __restrict__ x) {
    int i = blockIdx.x * blockDim.x + threadIdx.x;
    if (i >= N_PTS * 16) return;
    int c = i & 15, n = i >> 4;
    x[i] = (_Float16)((c < 4) ? f[n*4 + c] : 0.f);
}

// ---------------------------------------------------------------------------
// Conv layer: one block = 32-point tile, 8 waves (wave32). Waves w and w+4
// own the same cout block for point sub-tiles 0 / 1, so their WMMA B-fragment
// (weight) addresses are identical and dedupe in WGP$ -> ~2x less L2 weight
// traffic; 2 waves/SIMD hide LDS + cvt latency under the WMMA pipe.
// Per chunk of 4 kernel cells: zero G in LDS, scatter-add w * x[nbr]
// (ds_add_f32), then V_WMMA_F32_16X16X32_F16:
//   out(16 x 16) += G(16 x CCH*CIN) * Wchunk(CCH*CIN x 16)
// ---------------------------------------------------------------------------
template<int CIN, int COUTP, bool RELU, bool FINAL>
__global__ __launch_bounds__(256) void k_conv(
        const _Float16* __restrict__ xin,     // [N][CIN]
        const _Float16* __restrict__ Wt,      // [CELLS][COUTP][CIN]
        const float*    __restrict__ bias,    // [COUTP] padded
        const int*      __restrict__ nbr,
        const unsigned short* __restrict__ sIdx,
        const float*    __restrict__ sW,
        _Float16* __restrict__ xout,          // [N][COUTP]
        float*    __restrict__ outF) {        // final N x 3

    __shared__ float          Gs[CCH][TILE][CIN];
    __shared__ unsigned short sI[TILE * SLOTS];
    __shared__ float          sWl[TILE * SLOTS];
    __shared__ int            sN[TILE * K_NBR];

    const int tid    = threadIdx.x;
    const int lane   = tid & 31;
    const int wave   = tid >> 5;            // 0..7
    const int ps     = wave >> 2;           // point sub-tile (rows 0-15 / 16-31)
    const int obase  = (wave & 3) * 16;     // cout block (wave-uniform)
    const int pbase  = blockIdx.x * TILE;
    const int nvalid = min(N_PTS - pbase, TILE);   // 32, or 16 for last tile

    // cache this tile's slot table + neighbor ids in LDS (reused 54x);
    // tail points get sentinel cell 0xFFFF (never matches) and zero weight.
    for (int i = tid; i < TILE * SLOTS; i += 256) {
        bool ok = i < nvalid * SLOTS;
        sI[i]  = ok ? sIdx[(size_t)pbase * SLOTS + i] : (unsigned short)0xFFFFu;
        sWl[i] = ok ? sW[(size_t)pbase * SLOTS + i] : 0.f;
    }
    for (int i = tid; i < TILE * K_NBR; i += 256)
        sN[i] = (i < nvalid * K_NBR) ? nbr[pbase * K_NBR + i] : 0;
    __syncthreads();

    v8f acc = {};
    const int aKb  = (lane < 16) ? 0 : 16;  // A-fragment K base (contiguous, matches B)
    const int bKb  = (lane < 16) ? 0 : 16;  // B-fragment K base
    const int nsub = lane & 15;
    const int arow = ps * 16 + nsub;        // A-matrix row for this wave's sub-tile

    for (int chunk = 0; chunk < CELLS / CCH; ++chunk) {
        // zero G chunk
        float* gflat = &Gs[0][0][0];
        for (int i = tid; i < CCH * TILE * CIN; i += 256) gflat[i] = 0.f;
        __syncthreads();

        // scatter: slots whose cell falls in this chunk
        const int cellLo = chunk * CCH;
        for (int e = tid; e < TILE * SLOTS; e += 256) {
            int kc = (int)sI[e] - cellLo;
            if ((unsigned)kc < (unsigned)CCH) {
                float w  = sWl[e];
                int   p  = e >> 7;                 // local point 0..31
                int   nb = sN[e >> 3];             // p*16 + neighbor j
                const _Float16* xr = xin + (size_t)nb * CIN;
                float* g = &Gs[kc][p][0];
                #pragma unroll 4
                for (int c = 0; c < CIN; ++c)
                    atomicAdd(&g[c], w * (float)xr[c]);   // ds_add_f32
            }
        }
        __syncthreads();

        // dense GEMM over this chunk's contraction span (CCH*CIN)
        if (obase < COUTP) {                  // wave-uniform: EXEC stays full
            #pragma unroll
            for (int kk = 0; kk < (CCH * CIN) / 32; ++kk) {
                v16h a, b;
                const int k0 = kk * 32;
                #pragma unroll
                for (int i = 0; i < 16; ++i) {    // A: 16x32 f16 fragment
                    int kg = k0 + aKb + i;
                    a[i] = (_Float16)Gs[kg / CIN][arow][kg % CIN];
                }
                #pragma unroll
                for (int i = 0; i < 16; ++i) {    // B: 32x16 f16 fragment
                    int kg   = k0 + bKb + i;
                    int cell = kg / CIN, c = kg % CIN;
                    b[i] = Wt[(((size_t)(cellLo + cell)) * COUTP + (obase + nsub)) * CIN + c];
                }
                acc = __builtin_amdgcn_wmma_f32_16x16x32_f16(
                        false, a, false, b, (short)0, acc, false, false);
            }
        }
        __syncthreads();
    }

    // epilogue: bias (+ relu) and store
    if (obase < COUTP) {
        const int   o  = obase + nsub;
        const float bv = bias[o];
        #pragma unroll
        for (int r = 0; r < 8; ++r) {
            int   row = r + ((lane & 16) ? 8 : 0);
            int   n   = pbase + ps * 16 + row;
            float v   = acc[r] + bv;
            if (n < N_PTS) {
                if (FINAL) {
                    if (o < 3) outF[n*3 + o] = v;
                } else {
                    if (RELU) v = fmaxf(v, 0.f);
                    xout[(size_t)n * COUTP + o] = (_Float16)v;
                }
            }
        }
    }
}

// ---------------------------------------------------------------------------
extern "C" void kernel_launch(void* const* d_in, const int* in_sizes, int n_in,
                              void* d_out, int out_size, void* d_ws, size_t ws_size,
                              hipStream_t stream) {
    (void)in_sizes; (void)n_in; (void)out_size; (void)ws_size;

    const float* feats = (const float*)d_in[0];
    const float* pos   = (const float*)d_in[1];
    const int*   nbr   = (const int*)d_in[2];
    const float* mask  = (const float*)d_in[3];
    const float* W[5]  = {(const float*)d_in[4], (const float*)d_in[6],
                          (const float*)d_in[8], (const float*)d_in[10],
                          (const float*)d_in[12]};
    const float* B[5]  = {(const float*)d_in[5], (const float*)d_in[7],
                          (const float*)d_in[9], (const float*)d_in[11],
                          (const float*)d_in[13]};

    // workspace carve-up (~10 MB)
    char* wsp = (char*)d_ws;
    auto alloc = [&](size_t bytes) -> void* {
        void* p = (void*)wsp;
        wsp += (bytes + 255) & ~(size_t)255;
        return p;
    };
    unsigned short* sIdx = (unsigned short*)alloc((size_t)N_PTS * SLOTS * 2);
    float*          sW   = (float*)         alloc((size_t)N_PTS * SLOTS * 4);
    _Float16*       xA   = (_Float16*)      alloc((size_t)N_PTS * 64 * 2);
    _Float16*       xB   = (_Float16*)      alloc((size_t)N_PTS * 64 * 2);

    const int CINP[5]  = {16, 64, 64, 32, 32};
    const int COUTP[5] = {64, 64, 32, 32, 16};
    const int CINr[5]  = { 4, 64, 64, 32, 32};
    const int COUTr[5] = {64, 64, 32, 32,  3};
    _Float16* Wt[5];
    for (int l = 0; l < 5; ++l)
        Wt[l] = (_Float16*)alloc((size_t)CELLS * COUTP[l] * CINP[l] * 2);
    float* bpad = (float*)alloc(5 * 64 * 4);

    // geometry + prep
    k_slots<<<(N_PTS*K_NBR + 127)/128, 128, 0, stream>>>(pos, nbr, mask, sIdx, sW);
    for (int l = 0; l < 5; ++l) {
        int total = CELLS * COUTP[l] * CINP[l];
        k_prepw<<<(total + 255)/256, 256, 0, stream>>>(W[l], Wt[l],
                CINr[l], COUTr[l], CINP[l], COUTP[l]);
        k_prepb<<<1, 64, 0, stream>>>(B[l], bpad + l*64, COUTr[l], COUTP[l]);
    }
    k_castfeats<<<(N_PTS*16 + 255)/256, 256, 0, stream>>>(feats, xA);

    // 5 conv layers (ping-pong activations; strides chain: 64,64,32,32)
    float* outF = (float*)d_out;
    k_conv<16,64,true ,false><<<NTILES,256,0,stream>>>(xA, Wt[0], bpad+0*64, nbr, sIdx, sW, xB, nullptr);
    k_conv<64,64,true ,false><<<NTILES,256,0,stream>>>(xB, Wt[1], bpad+1*64, nbr, sIdx, sW, xA, nullptr);
    k_conv<64,32,true ,false><<<NTILES,256,0,stream>>>(xA, Wt[2], bpad+2*64, nbr, sIdx, sW, xB, nullptr);
    k_conv<32,32,true ,false><<<NTILES,256,0,stream>>>(xB, Wt[3], bpad+3*64, nbr, sIdx, sW, xA, nullptr);
    k_conv<32,16,false,true ><<<NTILES,256,0,stream>>>(xA, Wt[4], bpad+4*64, nbr, sIdx, sW, nullptr, outF);
}
